// IDGAT_27479200760359
// MI455X (gfx1250) — matrix operand order repaired and checked by reading, not compiled, not run on hardware
//
#include <hip/hip_runtime.h>
#include <hip/hip_bf16.h>
#include <math.h>

#define F 128
#define HEADS 8

typedef __attribute__((ext_vector_type(16))) __bf16 v16bf;
typedef __attribute__((ext_vector_type(8)))  float  v8f;
typedef __attribute__((ext_vector_type(4)))  unsigned int u32x4;
typedef __attribute__((ext_vector_type(8)))  unsigned int u32x8;

__device__ __forceinline__ __bf16 to_bf16(float f) { return (__bf16)f; }

__device__ __forceinline__ u32x8 make_g1(unsigned long long q0, unsigned long long q1,
                                         unsigned long long q2, unsigned long long q3) {
  u32x8 g;
  g[0] = (unsigned)q0; g[1] = (unsigned)(q0 >> 32);
  g[2] = (unsigned)q1; g[3] = (unsigned)(q1 >> 32);
  g[4] = (unsigned)q2; g[5] = (unsigned)(q2 >> 32);
  g[6] = (unsigned)q3; g[7] = (unsigned)(q3 >> 32);
  return g;
}

// TDM descriptor group0: count=1 (+gather flags), lds_addr, 57-bit global addr, type=2
__device__ __forceinline__ u32x4 make_g0(unsigned long long gaddr, unsigned int ldsoff,
                                         unsigned int flags) {
  u32x4 g;
  g[0] = 1u | flags;
  g[1] = ldsoff;
  g[2] = (unsigned)gaddr;
  g[3] = ((unsigned)(gaddr >> 32) & 0x1FFFFFFu) | (2u << 30);  // type=2 (image)
  return g;
}

// A-fragment: 16x32 bf16 (MxK) from an LDS-resident row. Lane L: M = L%16.
// VGPR0-3 hold K=0..7 (lanes 0-15) / K=8..15 (lanes 16-31); VGPR4-7: +16. (ISA 7.12.2)
__device__ __forceinline__ v16bf load_a_frag(const float* rowptr, int k0, int lane) {
  v16bf a;
  const int base = (lane < 16) ? 0 : 8;
#pragma unroll
  for (int v = 0; v < 8; ++v) {
    const int k = (v < 4) ? (base + 2 * v) : (16 + base + 2 * (v - 4));
    const float2 p = *(const float2*)(rowptr + k0 + k);
    a[2 * v]     = to_bf16(p.x);
    a[2 * v + 1] = to_bf16(p.y);
  }
  return a;
}

// B-fragment: 32x16 bf16 (KxN). Lane L: N = L%16; VGPR0-3 hold K=0..7 (lanes 0-15)
// / K=8..15 (lanes 16-31); VGPR4-7: +16. W row-major [128 x 128].
__device__ __forceinline__ v16bf load_b_frag(const float* __restrict__ W,
                                             int n0, int k0, int lane) {
  v16bf b;
  const int kbase = k0 + ((lane < 16) ? 0 : 8);
  const int col = n0 + (lane & 15);
#pragma unroll
  for (int v = 0; v < 8; ++v) {
    const int kk = (v < 4) ? 2 * v : 2 * v + 8;
    b[2 * v]     = to_bf16(W[(kbase + kk)     * F + col]);
    b[2 * v + 1] = to_bf16(W[(kbase + kk + 1) * F + col]);
  }
  return b;
}

// ---------------- Kernel 0: init m=-inf, denom=0, acc=0 ----------------
__global__ void idgat_init(float* __restrict__ m, float* __restrict__ denom,
                           float* __restrict__ acc, int nm, long nacc) {
  const long t = (long)blockIdx.x * blockDim.x + threadIdx.x;
  if (t < nm) { m[t] = -INFINITY; denom[t] = 0.0f; }
  if (t < nacc) acc[t] = 0.0f;
}

// ---------------- Kernel 1: Q/K/V GEMMs via TDM + WMMA bf16 ----------------
// grid.x = ceil(N/16), grid.y = 3 (0=Q,1=K,2=V); block = 256 = 8 wave32s.
// Wave 0 DMAs the 16x128 f32 x-tile into LDS via the Tensor Data Mover; all
// 8 waves then run one 16x16 column tile each off the shared A operand.
__global__ void __launch_bounds__(256)
idgat_qkv_wmma(const float* __restrict__ x,
               const float* __restrict__ wq, const float* __restrict__ bq,
               const float* __restrict__ wk, const float* __restrict__ bk,
               const float* __restrict__ wv,
               float* __restrict__ Q, float* __restrict__ K, float* __restrict__ V,
               int N) {
  __shared__ float xs[16 * F];                       // 8 KB tile
  const int lane = threadIdx.x & 31;
  const int wave = threadIdx.x >> 5;
  const int m0 = blockIdx.x * 16;
  const int n0 = wave * 16;
  const int mat = blockIdx.y;
  const float* W = (mat == 0) ? wq : (mat == 1) ? wk : wv;

  if (wave == 0) {
    const unsigned long long gaddr =
        (unsigned long long)(uintptr_t)(x + (size_t)m0 * F);
    const unsigned int ldsoff = (unsigned int)(uintptr_t)xs;
    int rows = N - m0; if (rows > 16) rows = 16;
    const u32x4 g0 = make_g0(gaddr, ldsoff, 0u);
    // group1: data_size=4B(2), tensor_dim0=128, tensor_dim1=rows,
    //         tile_dim0=128, tile_dim1=16, tensor_dim0_stride=128
    const unsigned long long q0 = (2ull << 16) | (128ull << 48);
    const unsigned long long q1 =
        ((unsigned long long)(unsigned)rows << 16) | (128ull << 48);
    const unsigned long long q2 = 16ull | (128ull << 32);
    const u32x8 g1 = make_g1(q0, q1, q2, 0ull);
    asm volatile("tensor_load_to_lds %0, %1" :: "s"(g0), "s"(g1) : "memory");
    __builtin_amdgcn_s_wait_tensorcnt(0);
  }
  __syncthreads();

  const float* rowptr = &xs[(lane & 15) * F];
  v8f c = {};
#pragma unroll
  for (int k0 = 0; k0 < F; k0 += 32) {
    v16bf a = load_a_frag(rowptr, k0, lane);
    v16bf b = load_b_frag(W, n0, k0, lane);
    c = __builtin_amdgcn_wmma_f32_16x16x32_bf16(false, a, false, b,
                                                (short)0, c, false, false);
  }

  const int col = n0 + (lane & 15);
  const int rbase = m0 + ((lane < 16) ? 0 : 8);
  const float bv = (mat == 0) ? bq[col] : (mat == 1) ? bk[col] : 0.0f;
  float* OUT = (mat == 0) ? Q : (mat == 1) ? K : V;
#pragma unroll
  for (int i = 0; i < 8; ++i) {
    const int r = rbase + i;
    if (r < N) {
      float vv = c[i] + bv;
      if (mat != 2) vv = fmaxf(vv, 0.0f);           // ReLU for Q, K
      OUT[(long)r * F + col] = vv;
    }
  }
}

// ---------------- Kernel 2: h_id = x[id] @ Wid via TDM gather-mode ----------------
// TDM gather mode pulls the 16 non-contiguous x rows (16-bit row indices,
// N=50000 < 65536) into LDS in one DMA; WMMA consumes the compacted tile and
// the result is scatter-added into V.
__global__ void __launch_bounds__(256)
idgat_id_wmma(const float* __restrict__ x, const int* __restrict__ id_index,
              const float* __restrict__ wid, float* __restrict__ V,
              int num_id, int N) {
  __shared__ float xs[16 * F];
  const int lane = threadIdx.x & 31;
  const int wave = threadIdx.x >> 5;
  const int m0 = blockIdx.x * 16;
  const int n0 = wave * 16;

  if (wave == 0) {
    const unsigned long long gaddr = (unsigned long long)(uintptr_t)x;
    const unsigned int ldsoff = (unsigned int)(uintptr_t)xs;
    int nval = num_id - m0; if (nval > 16) nval = 16;
    // gather_mode=1 (bit31), 16-bit indices (bit30=0)
    const u32x4 g0 = make_g0(gaddr, ldsoff, 1u << 31);
    const unsigned long long q0 = (2ull << 16) | (128ull << 48);
    const unsigned long long q1 =
        ((unsigned long long)(unsigned)N << 16) | (128ull << 48);
    const unsigned long long q2 =
        (unsigned long long)(unsigned)nval | (128ull << 32);
    const u32x8 g1 = make_g1(q0, q1, q2, 0ull);
    unsigned int idxs[16];
#pragma unroll
    for (int i = 0; i < 16; ++i) {
      const int r = m0 + i;
      idxs[i] = (unsigned)id_index[(r < num_id) ? r : (num_id - 1)];
    }
    u32x4 g2, g3;
#pragma unroll
    for (int w = 0; w < 4; ++w) {
      g2[w] = (idxs[2 * w] & 0xFFFFu) | (idxs[2 * w + 1] << 16);
      g3[w] = (idxs[8 + 2 * w] & 0xFFFFu) | (idxs[8 + 2 * w + 1] << 16);
    }
    asm volatile("tensor_load_to_lds %0, %1, %2, %3"
                 :: "s"(g0), "s"(g1), "s"(g2), "s"(g3) : "memory");
    __builtin_amdgcn_s_wait_tensorcnt(0);
  }
  __syncthreads();

  const float* rowptr = &xs[(lane & 15) * F];
  v8f c = {};
#pragma unroll
  for (int k0 = 0; k0 < F; k0 += 32) {
    v16bf a = load_a_frag(rowptr, k0, lane);
    v16bf b = load_b_frag(wid, n0, k0, lane);
    c = __builtin_amdgcn_wmma_f32_16x16x32_bf16(false, a, false, b,
                                                (short)0, c, false, false);
  }
  const int col = n0 + (lane & 15);
  const int rbase = m0 + ((lane < 16) ? 0 : 8);
#pragma unroll
  for (int i = 0; i < 8; ++i) {
    const int rid = rbase + i;
    if (rid < num_id) {
      const int dst = id_index[rid];
      atomicAdd(&V[(long)dst * F + col], c[i]);     // id rows may repeat
    }
  }
}

// ---------------- Kernel 3: per-edge attention logits + segment max ----------------
__global__ void __launch_bounds__(256)
idgat_logits(const float* __restrict__ Q, const float* __restrict__ K,
             const int* __restrict__ ei, float* __restrict__ att,
             float* __restrict__ mbuf, int E, int EP) {
  const int lane = threadIdx.x & 31;
  const int g = (blockIdx.x << 3) + (threadIdx.x >> 5);
  if (g >= EP) return;
  int row, col;
  if (g < E) { row = ei[g]; col = ei[E + g]; }
  else       { row = g - E; col = row; }            // self loop
  if (g + 64 < E) {                                 // gfx1250 global_prefetch_b8
    __builtin_prefetch(&ei[g + 64], 0, 1);
    __builtin_prefetch(&ei[E + g + 64], 0, 1);
  }
  const float4 q = *(const float4*)(Q + (long)row * F + lane * 4);
  const float4 k = *(const float4*)(K + (long)col * F + lane * 4);
  float p = q.x * k.x + q.y * k.y + q.z * k.z + q.w * k.w;
  p += __shfl_xor(p, 1, 32);
  p += __shfl_xor(p, 2, 32);
  if ((lane & 3) == 0) {
    const int h = lane >> 2;
    const float a = p * 0.25f;                      // 1/sqrt(dh), dh = 16
    att[(long)g * HEADS + h] = a;
    float* maddr = &mbuf[(long)row * HEADS + h];    // float atomic-max trick
    if (a >= 0.0f) atomicMax((int*)maddr, __float_as_int(a));
    else           atomicMin((unsigned int*)maddr, __float_as_uint(a));
  }
}

// ---------------- Kernel 4: e = exp(att - m[row]); denom += e ----------------
__global__ void idgat_expsum(const int* __restrict__ ei, const float* __restrict__ mbuf,
                             float* __restrict__ att, float* __restrict__ denom,
                             int E, int EP) {
  const long t = (long)blockIdx.x * blockDim.x + threadIdx.x;
  if (t >= (long)EP * HEADS) return;
  const int g = (int)(t >> 3);
  const int h = (int)(t & 7);
  const int row = (g < E) ? ei[g] : (g - E);
  const float e = __expf(att[t] - mbuf[(long)row * HEADS + h]);
  att[t] = e;                                       // reuse buffer in place
  atomicAdd(&denom[(long)row * HEADS + h], e);
}

// ---------------- Kernel 5: acc[row] += e * V[col] ----------------
__global__ void __launch_bounds__(256)
idgat_aggregate(const int* __restrict__ ei, const float* __restrict__ att,
                const float* __restrict__ V, float* __restrict__ acc,
                int E, int EP) {
  const int lane = threadIdx.x & 31;
  const int g = (blockIdx.x << 3) + (threadIdx.x >> 5);
  if (g >= EP) return;
  int row, col;
  if (g < E) { row = ei[g]; col = ei[E + g]; }
  else       { row = g - E; col = row; }
  __builtin_prefetch(V + (long)col * F + lane * 4, 0, 1);
#pragma unroll
  for (int j = 0; j < 4; ++j) {
    const int d = lane + 32 * j;
    const float w = att[(long)g * HEADS + (d >> 4)];
    const float vv = V[(long)col * F + d];
    atomicAdd(&acc[(long)row * F + d], w * vv);
  }
}

// ---------------- Kernel 6: out = acc/denom + bias ----------------
__global__ void idgat_finalize(const float* __restrict__ acc, const float* __restrict__ denom,
                               const float* __restrict__ bias, float* __restrict__ out,
                               long total) {
  const long t = (long)blockIdx.x * blockDim.x + threadIdx.x;
  if (t >= total) return;
  const int c = (int)(t & (F - 1));
  const long n = t >> 7;
  out[t] = acc[t] / denom[n * HEADS + (c >> 4)] + bias[c];
}

extern "C" void kernel_launch(void* const* d_in, const int* in_sizes, int n_in,
                              void* d_out, int out_size, void* d_ws, size_t ws_size,
                              hipStream_t stream) {
  const float* x    = (const float*)d_in[0];
  const int*   ei   = (const int*)  d_in[1];
  const int*   idi  = (const int*)  d_in[2];
  const float* wq   = (const float*)d_in[3];
  const float* bq   = (const float*)d_in[4];
  const float* wk   = (const float*)d_in[5];
  const float* bk   = (const float*)d_in[6];
  const float* wv   = (const float*)d_in[7];
  const float* wid  = (const float*)d_in[8];
  const float* bias = (const float*)d_in[9];

  const int N      = in_sizes[0] / F;
  const int E      = in_sizes[1] / 2;
  const int NUM_ID = in_sizes[2];
  const int EP     = E + N;

  // workspace layout (floats)
  float* ws = (float*)d_ws;
  size_t off = 0;
  float* Q    = ws + off; off += (size_t)N * F;
  float* Kb   = ws + off; off += (size_t)N * F;
  float* V    = ws + off; off += (size_t)N * F;
  float* att  = ws + off; off += (size_t)EP * HEADS;
  float* mbuf = ws + off; off += (size_t)N * HEADS;
  float* den  = ws + off; off += (size_t)N * HEADS;
  float* acc  = ws + off; off += (size_t)N * F;
  (void)ws_size; (void)n_in; (void)out_size;

  const long nacc = (long)N * F;
  idgat_init<<<(int)((nacc + 255) / 256), 256, 0, stream>>>(mbuf, den, acc,
                                                            N * HEADS, nacc);
  {
    dim3 grid((N + 15) / 16, 3);
    idgat_qkv_wmma<<<grid, 256, 0, stream>>>(x, wq, bq, wk, bk, wv, Q, Kb, V, N);
  }
  {
    dim3 grid((NUM_ID + 15) / 16);
    idgat_id_wmma<<<grid, 256, 0, stream>>>(x, idi, wid, V, NUM_ID, N);
  }
  idgat_logits<<<(EP + 7) / 8, 256, 0, stream>>>(Q, Kb, ei, att, mbuf, E, EP);
  {
    const long total = (long)EP * HEADS;
    idgat_expsum<<<(int)((total + 255) / 256), 256, 0, stream>>>(ei, mbuf, att, den, E, EP);
  }
  idgat_aggregate<<<(EP + 7) / 8, 256, 0, stream>>>(ei, att, V, acc, E, EP);
  idgat_finalize<<<(int)((nacc + 255) / 256), 256, 0, stream>>>(acc, den, bias,
                                                                (float*)d_out, nacc);
}
